// LanguageModule_36928128811122
// MI455X (gfx1250) — compile-verified, matrix operand order checked by reading, not run on hardware
//
#include <hip/hip_runtime.h>
#include <hip/hip_bf16.h>

// ---------------------------------------------------------------------------
// Model dims (fixed by the reference)
// ---------------------------------------------------------------------------
#define BQ   32
#define SQ   2048
#define NTQ  1536
#define DQ   512
#define HQ   1024
#define LQ   4
#define NQ   4          // NUM_Q
#define CBQ  32         // codebook entries
#define MTOT (BQ * SQ)  // 65536 rows
#define PADK 40         // padded LDS row length (elements): 80B = 20 banks

typedef __attribute__((ext_vector_type(16))) __bf16 v16bf;
typedef __attribute__((ext_vector_type(8)))  __bf16 v8bf;
typedef __attribute__((ext_vector_type(8)))  float  v8f;

union Frag16 { v16bf v; v8bf h[2]; };

__device__ __forceinline__ unsigned short f2bf(float f) {
    unsigned int u = __float_as_uint(f);
    unsigned int r = u + 0x7FFFu + ((u >> 16) & 1u);   // round-to-nearest-even
    if ((u & 0x7F800000u) == 0x7F800000u) r = u;       // inf / nan: truncate
    return (unsigned short)(r >> 16);
}
__device__ __forceinline__ float bf2f(unsigned short h) {
    return __uint_as_float(((unsigned int)h) << 16);
}

// CDNA5 async global->LDS copy (ASYNCcnt-tracked, no VGPR round trip).
__device__ __forceinline__ void async_copy_b128(unsigned lds_off, const void* gaddr) {
    asm volatile("global_load_async_to_lds_b128 %0, %1, off"
                 :: "v"(lds_off), "v"(gaddr) : "memory");
}
__device__ __forceinline__ void wait_async_le4() {
    asm volatile("s_wait_asynccnt 0x4" ::: "memory");
}
__device__ __forceinline__ void wait_async_0() {
    asm volatile("s_wait_asynccnt 0x0" ::: "memory");
}
// Flat addresses of LDS carry the LDS byte offset in bits [31:0] (ISA 10.2).
__device__ __forceinline__ unsigned lds_off_of(const void* p) {
    return (unsigned)(unsigned long long)p;
}

// ---------------------------------------------------------------------------
// Utility kernels
// ---------------------------------------------------------------------------
__global__ void zero_f32(float* __restrict__ p, size_t n) {
    size_t i = (size_t)blockIdx.x * blockDim.x + threadIdx.x;
    if (i < n) p[i] = 0.0f;
}

// float [K][N] per layer  ->  bf16 [N][K] per layer (transposed once per call)
__global__ void cvt_transpose_kernel(const float* __restrict__ src,
                                     unsigned short* __restrict__ dst,
                                     int K, int N) {
    size_t i = (size_t)blockIdx.x * 256 + threadIdx.x;
    size_t total = (size_t)LQ * K * N;
    if (i >= total) return;
    int n = (int)(i % N);
    size_t t = i / N;
    int k = (int)(t % K);
    int l = (int)(t / K);
    dst[((size_t)l * N + n) * K + k] = f2bf(src[i]);
}

// x[b,s,d] = (s < NT) ? embed[text[b,s]+1, d] : 0   (token 0 is the pad id)
__global__ void embed_kernel(const int* __restrict__ text,
                             const float* __restrict__ emb,
                             float* __restrict__ x) {
    size_t i = (size_t)blockIdx.x * 256 + threadIdx.x;       // float4 granules
    const size_t total = (size_t)MTOT * DQ / 4;
    if (i >= total) return;
    size_t tok = i >> 7;                 // 128 float4 per token
    int dc = (int)(i & 127) << 2;
    int s  = (int)(tok & (SQ - 1));
    float4 v = make_float4(0.f, 0.f, 0.f, 0.f);
    if (s < NTQ) {
        int b  = (int)(tok >> 11);
        int id = text[(size_t)b * NTQ + s] + 1;
        v = *(const float4*)(emb + (size_t)id * DQ + dc);
    }
    *(float4*)(x + tok * DQ + dc) = v;
}

// ||cb_k||^2 for all 4*32 entries
__global__ void cbnorm_kernel(const float* __restrict__ cbk, float* __restrict__ cbn) {
    int i = threadIdx.x;
    if (i < NQ * CBQ) {
        const float* p = cbk + (size_t)i * DQ;
        float s = 0.f;
        for (int d = 0; d < DQ; d += 4) {
            float4 c = *(const float4*)(p + d);
            s += c.x * c.x + c.y * c.y + c.z * c.z + c.w * c.w;
        }
        cbn[i] = s;
    }
}

// ---------------------------------------------------------------------------
// Fused depthwise conv7 + LayerNorm(D).  One wave per token (wave32),
// each lane owns 16 channels (stride-32 -> coalesced), bf16 output.
// x fits in the 192MB L2, so halo re-reads hit cache.
// ---------------------------------------------------------------------------
__global__ __launch_bounds__(256)
void dwln_kernel(const float* __restrict__ x,
                 const float* __restrict__ cw,   // [7][D]
                 const float* __restrict__ cb,   // [D]
                 const float* __restrict__ g,    // [D]
                 const float* __restrict__ be,   // [D]
                 unsigned short* __restrict__ y) {
    int w    = threadIdx.x >> 5;
    int lane = threadIdx.x & 31;
    size_t tok = (size_t)blockIdx.x * 8 + w;
    int b = (int)(tok >> 11);
    int s = (int)(tok & (SQ - 1));

    float v[16];
    float sum = 0.f, sq = 0.f;
#pragma unroll
    for (int j = 0; j < 16; ++j) {
        int d = lane + 32 * j;
        float a = cb[d];
#pragma unroll
        for (int t = 0; t < 7; ++t) {
            int sp = s + t - 3;
            if (sp >= 0 && sp < SQ)
                a += cw[t * DQ + d] * x[((size_t)b * SQ + sp) * DQ + d];
        }
        v[j] = a; sum += a; sq += a * a;
    }
    for (int off = 16; off; off >>= 1) {
        sum += __shfl_down(sum, off, 32);
        sq  += __shfl_down(sq,  off, 32);
    }
    sum = __shfl(sum, 0, 32);
    sq  = __shfl(sq,  0, 32);
    float mu  = sum * (1.f / DQ);
    float var = sq * (1.f / DQ) - mu * mu;
    float rs  = rsqrtf(var + 1e-6f);
#pragma unroll
    for (int j = 0; j < 16; ++j) {
        int d = lane + 32 * j;
        y[tok * DQ + d] = f2bf((v[j] - mu) * rs * g[d] + be[d]);
    }
}

// ---------------------------------------------------------------------------
// WMMA bf16 GEMM, 128x128x32 block tile, 8 waves -> each wave 64x32 (4x2 frags)
// A: M x K bf16 row-major.  Wt: N x K bf16 (pre-transposed) -> A and B tiles
// share the same contiguous 16B chunk pattern and are filled with
// double-buffered global_load_async_to_lds_b128 (ASYNCcnt pipelined).
// MODE 1: H = gelu(A@W + b1), bf16 out + atomic sum of H^2 into gsq[b][n]
// MODE 2: X = A@W + b2 + residual, masked rows zeroed, f32 out (in place)
// ---------------------------------------------------------------------------
template <int MODE>
__global__ __launch_bounds__(256)
void gemm_wmma(const unsigned short* __restrict__ A,   // M x K bf16
               const unsigned short* __restrict__ Wt,  // N x K bf16 (transposed)
               const float* __restrict__ bias,         // N
               void* __restrict__ outp,
               const float* __restrict__ res,          // MODE 2 residual
               float* __restrict__ gsq,                // MODE 1 stats
               int M, int N, int K) {
    __shared__ unsigned short As[2][128 * PADK];       // 2 x 10240 B
    __shared__ unsigned short Bs[2][128 * PADK];       // 2 x 10240 B  (as [n][k])

    const int tid  = threadIdx.x;
    const int lane = tid & 31;
    const int wv   = tid >> 5;
    const int wm   = wv >> 2;          // 0..1
    const int wn   = wv & 3;           // 0..3
    const int bm   = blockIdx.x * 128;
    const int bn   = blockIdx.y * 128;

    v8f acc[4][2];
#pragma unroll
    for (int mi = 0; mi < 4; ++mi)
#pragma unroll
        for (int ni = 0; ni < 2; ++ni)
#pragma unroll
            for (int r = 0; r < 8; ++r) acc[mi][ni][r] = 0.0f;

    // per-thread tile-fill assignment: chunks c and c+256 of 512 (8 elem = 16B)
    const int rA0 = tid >> 2,          kA0 = (tid & 3) << 3;
    const int rA1 = (tid + 256) >> 2,  kA1 = ((tid + 256) & 3) << 3;
    const unsigned short* gA0 = A  + (size_t)(bm + rA0) * K + kA0;
    const unsigned short* gA1 = A  + (size_t)(bm + rA1) * K + kA1;
    const unsigned short* gB0 = Wt + (size_t)(bn + rA0) * K + kA0;
    const unsigned short* gB1 = Wt + (size_t)(bn + rA1) * K + kA1;

    unsigned lA0[2], lA1[2], lB0[2], lB1[2];
#pragma unroll
    for (int b = 0; b < 2; ++b) {
        lA0[b] = lds_off_of(&As[b][rA0 * PADK + kA0]);
        lA1[b] = lds_off_of(&As[b][rA1 * PADK + kA1]);
        lB0[b] = lds_off_of(&Bs[b][rA0 * PADK + kA0]);
        lB1[b] = lds_off_of(&Bs[b][rA1 * PADK + kA1]);
    }

    const int lr   = lane & 15;
    const int ksel = (lane >> 4) << 3;   // 0 or 8 (element offset into K chunk)
    const int T    = K >> 5;             // K/32 tiles

    // prologue: tile 0 into buffer 0
    async_copy_b128(lA0[0], gA0);
    async_copy_b128(lA1[0], gA1);
    async_copy_b128(lB0[0], gB0);
    async_copy_b128(lB1[0], gB1);

    for (int kt = 0; kt < T; ++kt) {
        const int cur = kt & 1;
        const size_t ko = (size_t)(kt + 1) << 5;
        if (kt + 1 < T) {                       // prefetch next tile
            async_copy_b128(lA0[cur ^ 1], gA0 + ko);
            async_copy_b128(lA1[cur ^ 1], gA1 + ko);
            async_copy_b128(lB0[cur ^ 1], gB0 + ko);
            async_copy_b128(lB1[cur ^ 1], gB1 + ko);
            wait_async_le4();                   // older 4 (current tile) done
        } else {
            wait_async_0();
        }
        __syncthreads();                        // everyone's fills visible

        Frag16 a[4], b[2];
#pragma unroll
        for (int mi = 0; mi < 4; ++mi) {
            const __bf16* base =
                ((const __bf16*)&As[cur][0]) + (wm * 64 + mi * 16 + lr) * PADK;
            a[mi].h[0] = *(const v8bf*)(base + ksel);
            a[mi].h[1] = *(const v8bf*)(base + 16 + ksel);
        }
#pragma unroll
        for (int ni = 0; ni < 2; ++ni) {
            const __bf16* base =
                ((const __bf16*)&Bs[cur][0]) + (wn * 32 + ni * 16 + lr) * PADK;
            b[ni].h[0] = *(const v8bf*)(base + ksel);
            b[ni].h[1] = *(const v8bf*)(base + 16 + ksel);
        }
#pragma unroll
        for (int mi = 0; mi < 4; ++mi)
#pragma unroll
            for (int ni = 0; ni < 2; ++ni)
                acc[mi][ni] = __builtin_amdgcn_wmma_f32_16x16x32_bf16(
                    false, a[mi].v, false, b[ni].v,
                    (short)0, acc[mi][ni], false, false);

        __syncthreads();                        // reads done before next overwrite
    }

    // Epilogue.  C layout: element r of lane l -> m = r + 8*(l/16), n = l%16.
    const int mh = (lane >> 4) * 8;
    if (MODE == 1) {
        unsigned short* H = (unsigned short*)outp;
        const int batch = bm / SQ;                    // 128 | 2048, so constant
#pragma unroll
        for (int ni = 0; ni < 2; ++ni) {
            int n = bn + wn * 32 + ni * 16 + lr;
            float bi = bias[n];
            float ssq = 0.f;
#pragma unroll
            for (int mi = 0; mi < 4; ++mi) {
#pragma unroll
                for (int r = 0; r < 8; ++r) {
                    int m = bm + wm * 64 + mi * 16 + mh + r;
                    float v = acc[mi][ni][r] + bi;
                    float gv = 0.5f * v * (1.0f + erff(v * 0.70710678f));  // exact gelu
                    H[(size_t)m * N + n] = f2bf(gv);
                    ssq += gv * gv;
                }
            }
            atomicAdd(&gsq[batch * N + n], ssq);
        }
    } else {
        float* X = (float*)outp;
#pragma unroll
        for (int ni = 0; ni < 2; ++ni) {
            int n = bn + wn * 32 + ni * 16 + lr;
            float bi = bias[n];
#pragma unroll
            for (int mi = 0; mi < 4; ++mi) {
#pragma unroll
                for (int r = 0; r < 8; ++r) {
                    int m = bm + wm * 64 + mi * 16 + mh + r;
                    float v = acc[mi][ni][r] + bi + res[(size_t)m * N + n];
                    if ((m & (SQ - 1)) >= NTQ) v = 0.0f;   // padding mask
                    X[(size_t)m * N + n] = v;
                }
            }
        }
    }
}

// ---------------------------------------------------------------------------
// GRN stats: gx = sqrt(gsq), nx = gx / (mean_h(gx) + 1e-6).  One block per b.
// ---------------------------------------------------------------------------
__global__ __launch_bounds__(256)
void grn_stats_kernel(const float* __restrict__ gsq, float* __restrict__ nx) {
    __shared__ float red[256];
    int b = blockIdx.x, t = threadIdx.x;
    float gx[4];
    float part = 0.f;
#pragma unroll
    for (int j = 0; j < 4; ++j) {
        gx[j] = sqrtf(gsq[(size_t)b * HQ + t + 256 * j]);
        part += gx[j];
    }
    red[t] = part;
    __syncthreads();
    for (int off = 128; off; off >>= 1) {
        if (t < off) red[t] += red[t + off];
        __syncthreads();
    }
    float mean = red[0] * (1.f / HQ);
#pragma unroll
    for (int j = 0; j < 4; ++j)
        nx[(size_t)b * HQ + t + 256 * j] = gx[j] / (mean + 1e-6f);
}

// x = gg*(h*nx) + gb + h,  in place on bf16 h
__global__ __launch_bounds__(256)
void grn_apply_kernel(unsigned short* __restrict__ h,
                      const float* __restrict__ nx,
                      const float* __restrict__ gg,
                      const float* __restrict__ gb) {
    size_t i = (size_t)blockIdx.x * 256 + threadIdx.x;
    const size_t total = (size_t)MTOT * HQ;
    if (i >= total) return;
    int n = (int)(i & (HQ - 1));
    size_t tok = i >> 10;
    int b = (int)(tok >> 11);
    float hv = bf2f(h[i]);
    h[i] = f2bf(gg[n] * (hv * nx[(size_t)b * HQ + n]) + gb[n] + hv);
}

// ---------------------------------------------------------------------------
// RVQ: one wave per token; lane = codebook entry; residual staged in LDS.
// Codebooks (256KB) live in L2.  Writes z_q (f32), indices (as float),
// accumulates loss sum.
// ---------------------------------------------------------------------------
__global__ __launch_bounds__(256)
void rvq_kernel(const float* __restrict__ x,
                const float* __restrict__ cbk,   // [4][32][512]
                const float* __restrict__ cbn,   // [4][32]
                float* __restrict__ zq_out,
                float* __restrict__ idx_out,
                float* __restrict__ loss_acc) {
    __shared__ float rbuf[8 * DQ];
    int w    = threadIdx.x >> 5;
    int lane = threadIdx.x & 31;
    size_t tok = (size_t)blockIdx.x * 8 + w;

    const float* xr = x + tok * DQ;
    float r[16];
#pragma unroll
    for (int j = 0; j < 16; ++j) {
        int d = lane + 32 * j;
        r[j] = xr[d];
        rbuf[w * DQ + d] = r[j];
    }
    __syncthreads();

    float losssum = 0.f;
    int myidx = 0;
    for (int q = 0; q < NQ; ++q) {
        // distance to entry `lane`:  ||cb||^2 - 2 r.cb   (||r||^2 drops from argmin)
        const float* cbp = cbk + ((size_t)q * CBQ + lane) * DQ;
        const float* rp  = rbuf + w * DQ;
        float dot = 0.f;
        for (int d = 0; d < DQ; d += 4) {
            float4 c  = *(const float4*)(cbp + d);
            float4 rr = *(const float4*)(rp + d);
            dot += c.x * rr.x + c.y * rr.y + c.z * rr.z + c.w * rr.w;
        }
        float dist = cbn[q * CBQ + lane] - 2.f * dot;
        int bi = lane;
        for (int off = 16; off; off >>= 1) {
            float od = __shfl_down(dist, off, 32);
            int   oi = __shfl_down(bi,   off, 32);
            if (od < dist || (od == dist && oi < bi)) { dist = od; bi = oi; }
        }
        bi = __shfl(bi, 0, 32);
        if (lane == q) myidx = bi;

        // residual update + loss term:  (quant - r_old)^2 == r_new^2
        const float* qv = cbk + ((size_t)q * CBQ + bi) * DQ;
        float sq = 0.f;
#pragma unroll
        for (int j = 0; j < 16; ++j) {
            int d = lane + 32 * j;
            r[j] -= qv[d];
            sq += r[j] * r[j];
            rbuf[w * DQ + d] = r[j];
        }
        __syncthreads();
        for (int off = 16; off; off >>= 1) sq += __shfl_down(sq, off, 32);
        if (lane == 0) losssum += sq;
    }
    if (lane == 0) atomicAdd(loss_acc, losssum);

    int s = (int)(tok & (SQ - 1));
    if (lane < NQ) idx_out[tok * NQ + lane] = (float)myidx;
    bool masked = (s >= NTQ);
#pragma unroll
    for (int j = 0; j < 16; ++j) {
        int d = lane + 32 * j;
        zq_out[tok * DQ + d] = masked ? 0.0f : (xr[d] - r[j]);   // z_q = x - r_final
    }
}

__global__ void finalize_loss(const float* __restrict__ acc, float* __restrict__ out) {
    if (threadIdx.x == 0)
        out[0] = acc[0] * (1.25f / (4.0f * (float)MTOT * (float)DQ));
}

// ---------------------------------------------------------------------------
// Host launcher
// ---------------------------------------------------------------------------
extern "C" void kernel_launch(void* const* d_in, const int* in_sizes, int n_in,
                              void* d_out, int out_size, void* d_ws, size_t ws_size,
                              hipStream_t stream) {
    const int*   text  = (const int*)d_in[0];
    const float* emb   = (const float*)d_in[2];
    const float* dw_w  = (const float*)d_in[3];
    const float* dw_b  = (const float*)d_in[4];
    const float* ln_g  = (const float*)d_in[5];
    const float* ln_b  = (const float*)d_in[6];
    const float* w1    = (const float*)d_in[7];
    const float* b1    = (const float*)d_in[8];
    const float* grn_g = (const float*)d_in[9];
    const float* grn_b = (const float*)d_in[10];
    const float* w2    = (const float*)d_in[11];
    const float* b2    = (const float*)d_in[12];
    const float* cbk   = (const float*)d_in[13];

    // workspace carve-out
    char* ws = (char*)d_ws;
    float*          x    = (float*)ws;                       ws += (size_t)MTOT * DQ * 4;   // 134 MB
    unsigned short* ybf  = (unsigned short*)ws;              ws += (size_t)MTOT * DQ * 2;   //  67 MB
    unsigned short* hbf  = (unsigned short*)ws;              ws += (size_t)MTOT * HQ * 2;   // 134 MB
    unsigned short* w1t  = (unsigned short*)ws;              ws += (size_t)LQ * DQ * HQ * 2; // [l][H][D]
    unsigned short* w2t  = (unsigned short*)ws;              ws += (size_t)LQ * HQ * DQ * 2; // [l][D][H]
    float*          gsq  = (float*)ws;                       ws += (size_t)BQ * HQ * 4;
    float*          nx   = (float*)ws;                       ws += (size_t)BQ * HQ * 4;
    float*          cbn  = (float*)ws;                       ws += 256;
    float*          lacc = (float*)ws;                       ws += 256;

    float* out    = (float*)d_out;
    float* zq_out = out;
    float* l_out  = out + (size_t)MTOT * DQ;
    float* i_out  = l_out + 1;

    // weight conversion + transpose to bf16 [N][K] (once per call)
    {
        size_t n = (size_t)LQ * DQ * HQ;
        unsigned g = (unsigned)((n + 255) / 256);
        cvt_transpose_kernel<<<g, 256, 0, stream>>>(w1, w1t, DQ, HQ);  // -> [l][H][D]
        cvt_transpose_kernel<<<g, 256, 0, stream>>>(w2, w2t, HQ, DQ);  // -> [l][D][H]
    }
    cbnorm_kernel<<<1, 128, 0, stream>>>(cbk, cbn);

    // embedding gather + mask
    embed_kernel<<<(unsigned)(((size_t)MTOT * DQ / 4 + 255) / 256), 256, 0, stream>>>(text, emb, x);

    for (int l = 0; l < LQ; ++l) {
        dwln_kernel<<<MTOT / 8, 256, 0, stream>>>(
            x, dw_w + (size_t)l * 7 * DQ, dw_b + (size_t)l * DQ,
            ln_g + (size_t)l * DQ, ln_b + (size_t)l * DQ, ybf);

        zero_f32<<<(BQ * HQ + 255) / 256, 256, 0, stream>>>(gsq, (size_t)BQ * HQ);

        gemm_wmma<1><<<dim3(MTOT / 128, HQ / 128), 256, 0, stream>>>(
            ybf, w1t + (size_t)l * DQ * HQ, b1 + (size_t)l * HQ,
            (void*)hbf, nullptr, gsq, MTOT, HQ, DQ);

        grn_stats_kernel<<<BQ, 256, 0, stream>>>(gsq, nx);

        grn_apply_kernel<<<(unsigned)((size_t)MTOT * HQ / 256), 256, 0, stream>>>(
            hbf, nx, grn_g + (size_t)l * HQ, grn_b + (size_t)l * HQ);

        gemm_wmma<2><<<dim3(MTOT / 128, DQ / 128), 256, 0, stream>>>(
            hbf, w2t + (size_t)l * HQ * DQ, b2 + (size_t)l * DQ,
            (void*)x, x, nullptr, MTOT, DQ, HQ);
    }

    zero_f32<<<1, 256, 0, stream>>>(lacc, 1);
    rvq_kernel<<<MTOT / 8, 256, 0, stream>>>(x, cbk, cbn, zq_out, i_out, lacc);
    finalize_loss<<<1, 32, 0, stream>>>(lacc, l_out);
}
